// RVQVAE_67688684585449
// MI455X (gfx1250) — matrix operand order repaired and checked by reading, not compiled, not run on hardware
//
#include <hip/hip_runtime.h>
#include <hip/hip_bf16.h>

// ---------------------------------------------------------------------------
// RVQVAE forward for MI455X (gfx1250, wave32, WMMA + TDM).
// All convolutions and the VQ score matrices map onto one implicit-im2col
// GEMM kernel built around v_wmma_f32_16x16x32_f16. Weight tiles are staged
// to LDS by the Tensor Data Mover (tensor_load_to_lds, TENSORcnt); the
// activations are staged with a hoisted, vectorized im2col. f16 storage,
// f32 accumulation, f32 epilogues.
// ---------------------------------------------------------------------------

typedef __attribute__((ext_vector_type(16))) _Float16 v16h;
typedef __attribute__((ext_vector_type(8)))  _Float16 v8h;
typedef __attribute__((ext_vector_type(4)))  _Float16 v4h;
typedef __attribute__((ext_vector_type(8)))  float    v8f;
typedef __attribute__((ext_vector_type(4)))  unsigned int u32x4;
typedef __attribute__((ext_vector_type(8)))  int          i32x8;
typedef __attribute__((ext_vector_type(4)))  int          i32x4;

#if defined(__AMDGCN__) && __has_builtin(__builtin_amdgcn_tensor_load_to_lds) && \
    __has_builtin(__builtin_amdgcn_s_wait_tensorcnt)
#define USE_TDM 1
#else
#define USE_TDM 0
#endif

#define FLAG_RELU_IN  1   // relu applied to activation operand as it is loaded
#define FLAG_RELU_OUT 2   // relu applied in epilogue (after bias)
#define FLAG_UPSAMPLE 4   // nearest-neighbor x2 upsample folded into im2col

#define TILE_M 64
#define TILE_N 128
#define TILE_K 32
#define LDA_P  40         // LDS row pitch (halfs) for A tile (64 x 32 + 8 pad)
#define LDB_P  40         // LDS col pitch (halfs) for B tile (transposed 128 x 32)

// model dims
#define BSZ    32
#define TFULL  512
#define THALF  256
#define BODY   393
#define WID    512
#define DWID   1024
#define NBCODE 1024
#define CDIM   512
#define NQUANT 6
#define OUTN   (BSZ*BODY*TFULL)   // 6,438,912

// ---------------------------------------------------------------------------
// WMMA implicit-im2col GEMM.
//   out[b,o,t] = bias[o] + sum_{i,r} wgt[o, i*Kw+r] * act[b, i, src(t,r)]
// M = Cout, N = B*Tout (n = b*Tout + t), K = Cin*Kw (padded to Kpad % 32 == 0).
// Tin/Tout are powers of two (shift addressing). Block: 256 thr = 8 waves,
// tile 64x128; each wave computes 32x32 via 2x2 v_wmma_f32_16x16x32_f16.
// ---------------------------------------------------------------------------
__global__ void __launch_bounds__(256)
wmma_conv_gemm(const _Float16* __restrict__ wgt,    // (M, Kpad) f16
               const _Float16* __restrict__ act,    // (B, Cin, Tin) f16
               const float*    __restrict__ bias,   // (M) or nullptr
               const _Float16* __restrict__ res16,  // (B, M, Tout) or nullptr
               _Float16*       __restrict__ out16,  // (B, M, Tout) or nullptr
               float*          __restrict__ outf,   // (B, M, Tout) or nullptr
               int M, int Nn, int Kreal, int Kpad,
               int Cin, int Tin, int tinShift, int toutShift,
               int Kw, int stride, int pad, int dil, int flags)
{
    __shared__ _Float16 lA [TILE_M * LDA_P];
    __shared__ _Float16 lBT[TILE_N * LDB_P];

    const int tid    = threadIdx.x;
    const int lane   = tid & 31;
    const int wave   = tid >> 5;          // 0..7
    const int waveM  = wave >> 2;         // 0..1 -> 32 rows each
    const int waveN  = wave & 3;          // 0..3 -> 32 cols each
    const int blockM = blockIdx.y * TILE_M;
    const int blockN = blockIdx.x * TILE_N;

    const int  Tout   = 1 << toutShift;
    const bool reluIn = (flags & FLAG_RELU_IN)  != 0;
    const bool upsamp = (flags & FLAG_UPSAMPLE) != 0;

    v8f acc[2][2];
    for (int mi = 0; mi < 2; ++mi)
        for (int nj = 0; nj < 2; ++nj)
            for (int e = 0; e < 8; ++e) acc[mi][nj][e] = 0.0f;

#if !USE_TDM
    // manual A loader: thread -> (row 0..63, 8-half chunk 0..3), aligned b128
    const int aRow   = tid >> 2;
    const int aChunk = (tid & 3) * 8;
#endif
    // B loader: thread -> 4x4 patch: 4 consecutive k, 4 consecutive n.
    // blockN is a multiple of 128 and Tout >= 128, so a 4-wide n-run never
    // crosses a batch boundary -> hoist b/t0 out of the element loops.
    const int bKK4  = (tid & 7) * 4;       // k offset within tile: 0..28
    const int bCol4 = (tid >> 3) * 4;      // n offset within tile: 0..124
    const int n0    = blockN + bCol4;
    const int bb    = n0 >> toutShift;
    const int t0    = n0 & (Tout - 1);
    const size_t chanBase = ((size_t)bb * Cin) << tinShift;

    for (int k0 = 0; k0 < Kpad; k0 += TILE_K) {
#if USE_TDM
        // ---- stage A (weights) tile via Tensor Data Mover -----------------
        // 2D D#: tile 32(k) x 64(m) halfs, row stride Kpad, +16B/row LDS pad
        // (pad_interval code 3 = 16 DWORDs, pad_amount code 3 = 4 DWORDs)
        // to reproduce the LDA_P=40-half pitch. M-edge rows zero-fill in HW.
        if (wave == 0) {
            const unsigned long long ga =
                (unsigned long long)(const void*)wgt +
                (((size_t)blockM * (size_t)Kpad + (size_t)k0) * 2ull);
            const unsigned td0 = (unsigned)(Kpad - k0);     // remaining K
            const unsigned td1 = (unsigned)(M - blockM);    // remaining rows
            u32x4 g0;
            g0[0] = 1u;                                     // count=1, load D#
            g0[1] = (unsigned)(size_t)(const void*)lA;      // LDS byte address
            g0[2] = (unsigned)(ga & 0xffffffffull);
            g0[3] = (unsigned)((ga >> 32) & 0x1ffffffull) | (2u << 30); // type=2
            i32x8 g1;
            g1[0] = (int)((1u << 16) | (1u << 20) | (3u << 22) | (3u << 25));
            g1[1] = (int)((td0 & 0xffffu) << 16);           // tensor_dim0 lo
            g1[2] = (int)((td0 >> 16) | ((td1 & 0xffffu) << 16));
            g1[3] = (int)((td1 >> 16) | (32u << 16));       // tile_dim0 = 32
            g1[4] = 64;                                     // tile_dim1 = 64
            g1[5] = Kpad;                                   // dim0 stride lo
            g1[6] = 0;                                      // stride hi, dim1 stride
            g1[7] = 0;
            i32x4 gz4 = {0, 0, 0, 0};
            i32x8 gz8 = {0, 0, 0, 0, 0, 0, 0, 0};
            __builtin_amdgcn_tensor_load_to_lds(g0, g1, gz4, gz4, gz8, 0);
        }
#else
        // ---- stage A (weights) tile: 64 x 32 f16, aligned b128 loads ------
        {
            uint4 v = {0u, 0u, 0u, 0u};
            const int gm = blockM + aRow;
            if (gm < M) {
                const _Float16* gp = wgt + (size_t)gm * Kpad + k0 + aChunk;
                v = *(const uint4*)gp;
                if (k0 + TILE_K < Kpad)
                    __builtin_prefetch(gp + TILE_K, 0, 1);  // global_prefetch_b8
            }
            *(uint4*)(lA + aRow * LDA_P + aChunk) = v;
        }
#endif
        // ---- stage B (implicit im2col) transposed: lBT[n][k], b64 stores ---
        {
            v4h vreg[4];
            #pragma unroll
            for (int jn = 0; jn < 4; ++jn)
                #pragma unroll
                for (int e = 0; e < 4; ++e) vreg[jn][e] = (_Float16)0.0f;

            int k = k0 + bKK4;
            int ci = k / Kw;               // single division per k-step
            int cr = k - ci * Kw;
            #pragma unroll
            for (int kidx = 0; kidx < 4; ++kidx) {
                if (k + kidx < Kreal) {
                    const int rd = cr * dil - pad;
                    const size_t rowBase = chanBase + ((size_t)ci << tinShift);
                    #pragma unroll
                    for (int jn = 0; jn < 4; ++jn) {
                        if (n0 + jn >= Nn) continue;
                        const int t = t0 + jn;
                        _Float16 v = (_Float16)0.0f;
                        if (upsamp) {
                            const int ts = t + rd;          // virtual len 2*Tin
                            if ((unsigned)ts < (unsigned)(Tin << 1))
                                v = act[rowBase + (ts >> 1)];
                        } else {
                            const int ts = t * stride + rd;
                            if ((unsigned)ts < (unsigned)Tin)
                                v = act[rowBase + ts];
                        }
                        if (reluIn && (float)v < 0.0f) v = (_Float16)0.0f;
                        vreg[jn][kidx] = v;
                    }
                }
                if (++cr == Kw) { cr = 0; ++ci; }
            }
            #pragma unroll
            for (int jn = 0; jn < 4; ++jn)
                *(v4h*)(lBT + (bCol4 + jn) * LDB_P + bKK4) = vreg[jn];
        }
#if USE_TDM
        if (wave == 0) __builtin_amdgcn_s_wait_tensorcnt(0);
#endif
        __syncthreads();

        // ---- gather fragments per ISA 7.12.2 layouts, then 4 WMMAs ----
        v16h afr[2], bfr[2];
        {
            const int rsel = lane & 15;
            const int kbA  = (lane < 16) ? 0 : 8;   // A: lanes0-15 K{0..7,16..23}
            for (int mi = 0; mi < 2; ++mi) {
                const _Float16* p = lA + (waveM * 32 + mi * 16 + rsel) * LDA_P + kbA;
                v8h lo = *(const v8h*)p;
                v8h hi = *(const v8h*)(p + 16);
                for (int e = 0; e < 8; ++e) { afr[mi][e] = lo[e]; afr[mi][e + 8] = hi[e]; }
            }
            const int kbB = (lane < 16) ? 0 : 8;    // B: lanes0-15 K{0..7,16..23}
            for (int nj = 0; nj < 2; ++nj) {
                const _Float16* p = lBT + (waveN * 32 + nj * 16 + rsel) * LDB_P + kbB;
                v8h lo = *(const v8h*)p;
                v8h hi = *(const v8h*)(p + 16);
                for (int e = 0; e < 8; ++e) { bfr[nj][e] = lo[e]; bfr[nj][e + 8] = hi[e]; }
            }
        }
        for (int mi = 0; mi < 2; ++mi)
            for (int nj = 0; nj < 2; ++nj)
                acc[mi][nj] = __builtin_amdgcn_wmma_f32_16x16x32_f16(
                    false, afr[mi], false, bfr[nj], (short)0, acc[mi][nj], false, false);

        __syncthreads();
    }

    // ---- epilogue: bias + relu + residual-add, dual f32/f16 stores ----
    const bool reluOut = (flags & FLAG_RELU_OUT) != 0;
    const int  rowOff  = (lane >> 4) * 8;   // C layout: VGPR r -> M = r + 8*(lane>=16)
    const int  colOff  = lane & 15;
    for (int mi = 0; mi < 2; ++mi)
        for (int nj = 0; nj < 2; ++nj) {
            const int n = blockN + waveN * 32 + nj * 16 + colOff;
            if (n >= Nn) continue;
            const int b = n >> toutShift;
            const int t = n & (Tout - 1);
            for (int r = 0; r < 8; ++r) {
                const int o = blockM + waveM * 32 + mi * 16 + rowOff + r;
                if (o >= M) continue;
                float v = acc[mi][nj][r];
                if (bias) v += bias[o];
                if (reluOut) v = fmaxf(v, 0.0f);
                const size_t ix = (((size_t)b * M + o) << toutShift) + t;
                if (res16) v += (float)res16[ix];
                if (out16) out16[ix] = (_Float16)v;
                if (outf)  outf [ix] = v;
            }
        }
}

// ---------------------------------------------------------------------------
// Small helper kernels
// ---------------------------------------------------------------------------
__global__ void pack_weight(const float* __restrict__ w, _Float16* __restrict__ o,
                            int O, int K, int Kpad)
{
    int i = blockIdx.x * blockDim.x + threadIdx.x;
    int total = O * Kpad;
    if (i >= total) return;
    int row = i / Kpad, k = i - row * Kpad;
    o[i] = (k < K) ? (_Float16)w[(size_t)row * K + k] : (_Float16)0.0f;
}

__global__ void f32_to_f16(const float* __restrict__ in, _Float16* __restrict__ out, int n)
{
    int i = blockIdx.x * blockDim.x + threadIdx.x;
    if (i < n) out[i] = (_Float16)in[i];
}

// x_body (B, T, BODY) f32 -> x16 (B, BODY, T) f16
__global__ void pack_xin(const float* __restrict__ xb, _Float16* __restrict__ o)
{
    int i = blockIdx.x * blockDim.x + threadIdx.x;
    if (i >= OUTN) return;
    int t = i & (TFULL - 1);
    int rest = i >> 9;                 // TFULL == 512
    int c = rest % BODY;
    int b = rest / BODY;
    o[i] = (_Float16)xb[((size_t)b * TFULL + t) * BODY + c];
}

__global__ void fill_f32(float* p, float v, int n)
{ int i = blockIdx.x * blockDim.x + threadIdx.x; if (i < n) p[i] = v; }

__global__ void fill_i32(int* p, int v, int n)
{ int i = blockIdx.x * blockDim.x + threadIdx.x; if (i < n) p[i] = v; }

__global__ void cnorm_kernel(const float* __restrict__ cb, float* __restrict__ cnorm,
                             int NB, int D)
{
    int c = blockIdx.x * blockDim.x + threadIdx.x;
    if (c >= NB) return;
    float s = 0.0f;
    for (int d = 0; d < D; ++d) { float v = cb[(size_t)c * D + d]; s += v * v; }
    cnorm[c] = s;
}

// One block per (b,t): argmin_c (||c||^2 - 2*score) then ST-update of
// quant/residual, commit-loss partial sum, code histogram.
__global__ void __launch_bounds__(256)
vq_assign_update(const float* __restrict__ scores,   // (B, NB, Tq)
                 const float* __restrict__ cnorm,    // (NB)
                 const float* __restrict__ cb,       // (NB, D) f32
                 float* __restrict__ residual,       // (B, D, Tq)
                 float* __restrict__ quant,          // (B, D, Tq)
                 int*   __restrict__ counts,         // (NB)
                 float* __restrict__ commit_acc,
                 int Tq, int NB, int D)
{
    __shared__ float sval[256];
    __shared__ int   sidx[256];
    const int pos = blockIdx.x;
    const int b = pos / Tq, t = pos - b * Tq;
    const int tid = threadIdx.x;

    float best = 3.4e38f; int bi = 0;
    for (int c = tid; c < NB; c += 256) {
        float d = cnorm[c] - 2.0f * scores[((size_t)b * NB + c) * Tq + t];
        if (d < best) { best = d; bi = c; }
    }
    sval[tid] = best; sidx[tid] = bi;
    __syncthreads();
    for (int s = 128; s > 0; s >>= 1) {
        if (tid < s) {
            float ov = sval[tid + s]; int oi = sidx[tid + s];
            if (ov < sval[tid] || (ov == sval[tid] && oi < sidx[tid])) {
                sval[tid] = ov; sidx[tid] = oi;
            }
        }
        __syncthreads();
    }
    const int idx = sidx[0];
    __syncthreads();

    float lsum = 0.0f;
    for (int c = tid; c < D; c += 256) {
        const float q = cb[(size_t)idx * D + c];
        const size_t ix = ((size_t)b * D + c) * Tq + t;
        const float r = residual[ix];
        const float diff = r - q;
        lsum += diff * diff;
        quant[ix] += q;         // forward value of the ST estimator
        residual[ix] = diff;
    }
    sval[tid] = lsum;
    __syncthreads();
    for (int s = 128; s > 0; s >>= 1) {
        if (tid < s) sval[tid] += sval[tid + s];
        __syncthreads();
    }
    if (tid == 0) {
        atomicAdd(commit_acc, sval[0]);
        atomicAdd(&counts[idx], 1);
    }
}

__global__ void __launch_bounds__(256)
vq_finalize(const int* __restrict__ counts, const float* __restrict__ commit_acc,
            float* __restrict__ commit_out, float* __restrict__ perp_out,
            int nq, int NB, float invNT, float invElems)
{
    __shared__ float sdata[256];
    const int tid = threadIdx.x;
    float perp = 0.0f;
    for (int l = 0; l < nq; ++l) {
        float s = 0.0f;
        for (int c = tid; c < NB; c += 256) {
            float p = (float)counts[l * NB + c] * invNT;
            s += p * logf(p + 1e-9f);
        }
        sdata[tid] = s;
        __syncthreads();
        for (int r = 128; r > 0; r >>= 1) {
            if (tid < r) sdata[tid] += sdata[tid + r];
            __syncthreads();
        }
        if (tid == 0) perp += expf(-sdata[0]);
        __syncthreads();
    }
    if (tid == 0) {
        *commit_out = commit_acc[0] * invElems;
        *perp_out   = perp / (float)nq;
    }
}

// ---------------------------------------------------------------------------
// Host-side orchestration
// ---------------------------------------------------------------------------
static void run_conv(hipStream_t s,
                     const _Float16* wgt, const _Float16* act, const float* bias,
                     const _Float16* res16, _Float16* out16, float* outf,
                     int M, int Bb, int Cin, int Tin, int Tout,
                     int Kw, int stride, int pad, int dil, int flags)
{
    const int Nn = Bb * Tout;
    const int Kreal = Cin * Kw;
    const int Kpad = (Kreal + 31) & ~31;
    dim3 grid((Nn + TILE_N - 1) / TILE_N, (M + TILE_M - 1) / TILE_M);
    wmma_conv_gemm<<<grid, 256, 0, s>>>(wgt, act, bias, res16, out16, outf,
                                        M, Nn, Kreal, Kpad,
                                        Cin, Tin, __builtin_ctz(Tin), __builtin_ctz(Tout),
                                        Kw, stride, pad, dil, flags);
}

extern "C" void kernel_launch(void* const* d_in, const int* in_sizes, int n_in,
                              void* d_out, int out_size, void* d_ws, size_t ws_size,
                              hipStream_t stream)
{
    (void)in_sizes; (void)n_in; (void)out_size; (void)ws_size;

    // Assumed flattening: setup_inputs() insertion order, recursively.
    const float* x_body   = (const float*)d_in[0];
    const float* e_w_in   = (const float*)d_in[1];
    const float* e_b_in   = (const float*)d_in[2];
    const float* e_w_down = (const float*)d_in[3];
    const float* e_b_down = (const float*)d_in[4];
    const float *e_rw1[3], *e_rb1[3], *e_rw2[3], *e_rb2[3];
    for (int i = 0; i < 3; ++i) {
        e_rw1[i] = (const float*)d_in[5 + 4 * i];
        e_rb1[i] = (const float*)d_in[6 + 4 * i];
        e_rw2[i] = (const float*)d_in[7 + 4 * i];
        e_rb2[i] = (const float*)d_in[8 + 4 * i];
    }
    const float* e_w_out  = (const float*)d_in[17];
    const float* e_b_out  = (const float*)d_in[18];
    const float* de_w_in  = (const float*)d_in[19];
    const float* de_b_in  = (const float*)d_in[20];
    const float *d_rw1[3], *d_rb1[3], *d_rw2[3], *d_rb2[3];
    for (int i = 0; i < 3; ++i) {
        d_rw1[i] = (const float*)d_in[21 + 4 * i];
        d_rb1[i] = (const float*)d_in[22 + 4 * i];
        d_rw2[i] = (const float*)d_in[23 + 4 * i];
        d_rb2[i] = (const float*)d_in[24 + 4 * i];
    }
    const float* de_w_up  = (const float*)d_in[33];
    const float* de_b_up  = (const float*)d_in[34];
    const float* de_w_mid = (const float*)d_in[35];
    const float* de_b_mid = (const float*)d_in[36];
    const float* de_w_out = (const float*)d_in[37];
    const float* de_b_out = (const float*)d_in[38];
    const float* cbs[NQUANT];
    for (int l = 0; l < NQUANT; ++l) cbs[l] = (const float*)d_in[39 + l];

    float* out_f = (float*)d_out;

    // workspace bump allocator (256B aligned)
    size_t off = 0;
    auto alloc = [&](size_t bytes) -> void* {
        size_t o = (off + 255) & ~(size_t)255;
        off = o + bytes;
        return (void*)((char*)d_ws + o);
    };
    auto packW = [&](const float* w, int O, int K) -> _Float16* {
        int Kp = (K + 31) & ~31;
        _Float16* p = (_Float16*)alloc((size_t)O * Kp * sizeof(_Float16));
        int total = O * Kp;
        pack_weight<<<(total + 255) / 256, 256, 0, stream>>>(w, p, O, K, Kp);
        return p;
    };

    // activation / VQ buffers
    _Float16* X16   = (_Float16*)alloc((size_t)OUTN * 2);                 // (B,393,512)
    _Float16* H1    = (_Float16*)alloc((size_t)BSZ * WID * TFULL * 2);    // (B,512,512)
    _Float16* H2a   = (_Float16*)alloc((size_t)BSZ * WID * THALF * 2);
    _Float16* H2b   = (_Float16*)alloc((size_t)BSZ * WID * THALF * 2);
    _Float16* T1    = (_Float16*)alloc((size_t)BSZ * WID * THALF * 2);
    float*    RESID = (float*)   alloc((size_t)BSZ * CDIM * THALF * 4);
    float*    QUANT = (float*)   alloc((size_t)BSZ * CDIM * THALF * 4);
    _Float16* R16   = (_Float16*)alloc((size_t)BSZ * CDIM * THALF * 2);
    float*    SCORE = (float*)   alloc((size_t)BSZ * NBCODE * THALF * 4);
    _Float16* XQ16  = (_Float16*)alloc((size_t)BSZ * CDIM * THALF * 2);
    _Float16* D1a   = (_Float16*)alloc((size_t)BSZ * DWID * THALF * 2);
    _Float16* D1b   = (_Float16*)alloc((size_t)BSZ * DWID * THALF * 2);
    _Float16* DT1   = (_Float16*)alloc((size_t)BSZ * DWID * THALF * 2);
    _Float16* D2    = (_Float16*)alloc((size_t)BSZ * DWID * TFULL * 2);
    _Float16* D3    = (_Float16*)alloc((size_t)BSZ * DWID * TFULL * 2);
    float*    CNORM = (float*)   alloc((size_t)NQUANT * NBCODE * 4);
    int*      CNT   = (int*)     alloc((size_t)NQUANT * NBCODE * 4);
    float*    CACC  = (float*)   alloc(256);

    // packed f16 weights
    _Float16* pe_w_in   = packW(e_w_in,   WID, BODY * 3);
    _Float16* pe_w_down = packW(e_w_down, WID, WID * 4);
    _Float16 *pe_rw1[3], *pe_rw2[3];
    for (int i = 0; i < 3; ++i) {
        pe_rw1[i] = packW(e_rw1[i], WID, WID * 3);
        pe_rw2[i] = packW(e_rw2[i], WID, WID);
    }
    _Float16* pe_w_out  = packW(e_w_out,  WID, WID * 3);
    _Float16* pd_w_in   = packW(de_w_in,  DWID, WID * 3);
    _Float16 *pd_rw1[3], *pd_rw2[3];
    for (int i = 0; i < 3; ++i) {
        pd_rw1[i] = packW(d_rw1[i], DWID, DWID * 3);
        pd_rw2[i] = packW(d_rw2[i], DWID, DWID);
    }
    _Float16* pd_w_up   = packW(de_w_up,  DWID, DWID * 3);
    _Float16* pd_w_mid  = packW(de_w_mid, DWID, DWID * 3);
    _Float16* pd_w_out  = packW(de_w_out, BODY, DWID * 3);
    _Float16* pcb[NQUANT];
    for (int l = 0; l < NQUANT; ++l) pcb[l] = packW(cbs[l], NBCODE, CDIM);

    // ---- input transpose + f16 ----
    pack_xin<<<(OUTN + 255) / 256, 256, 0, stream>>>(x_body, X16);

    // ---- encoder ----
    run_conv(stream, pe_w_in, X16, e_b_in, nullptr, H1, nullptr,
             WID, BSZ, BODY, TFULL, TFULL, 3, 1, 1, 1, FLAG_RELU_OUT);
    run_conv(stream, pe_w_down, H1, e_b_down, nullptr, H2a, nullptr,
             WID, BSZ, WID, TFULL, THALF, 4, 2, 1, 1, 0);
    _Float16* cur = H2a; _Float16* alt = H2b;
    for (int i = 0; i < 3; ++i) {
        int dil = 1; for (int j = 0; j < i; ++j) dil *= 3;   // 1,3,9
        run_conv(stream, pe_rw1[i], cur, e_rb1[i], nullptr, T1, nullptr,
                 WID, BSZ, WID, THALF, THALF, 3, 1, dil, dil, FLAG_RELU_IN);
        run_conv(stream, pe_rw2[i], T1, e_rb2[i], cur, alt, nullptr,
                 WID, BSZ, WID, THALF, THALF, 1, 1, 0, 1, FLAG_RELU_IN);
        _Float16* tmp = cur; cur = alt; alt = tmp;
    }
    run_conv(stream, pe_w_out, cur, e_b_out, nullptr, nullptr, RESID,
             CDIM, BSZ, WID, THALF, THALF, 3, 1, 1, 1, 0);   // x_enc -> residual

    // ---- residual VQ ----
    const int NZ = BSZ * CDIM * THALF;       // 4,194,304
    fill_f32<<<(NZ + 255) / 256, 256, 0, stream>>>(QUANT, 0.0f, NZ);
    fill_f32<<<1, 1, 0, stream>>>(CACC, 0.0f, 1);
    fill_i32<<<(NQUANT * NBCODE + 255) / 256, 256, 0, stream>>>(CNT, 0, NQUANT * NBCODE);
    for (int l = 0; l < NQUANT; ++l)
        cnorm_kernel<<<(NBCODE + 255) / 256, 256, 0, stream>>>(cbs[l], CNORM + l * NBCODE,
                                                               NBCODE, CDIM);
    for (int l = 0; l < NQUANT; ++l) {
        f32_to_f16<<<(NZ + 255) / 256, 256, 0, stream>>>(RESID, R16, NZ);
        // scores = residual @ cb^T as a 1x1 "conv":  M=1024, K=512, N=8192
        run_conv(stream, pcb[l], R16, nullptr, nullptr, nullptr, SCORE,
                 NBCODE, BSZ, CDIM, THALF, THALF, 1, 1, 0, 1, 0);
        vq_assign_update<<<BSZ * THALF, 256, 0, stream>>>(
            SCORE, CNORM + l * NBCODE, cbs[l], RESID, QUANT,
            CNT + l * NBCODE, CACC, THALF, NBCODE, CDIM);
    }
    f32_to_f16<<<(NZ + 255) / 256, 256, 0, stream>>>(QUANT, XQ16, NZ);

    // ---- decoder ----
    run_conv(stream, pd_w_in, XQ16, de_b_in, nullptr, D1a, nullptr,
             DWID, BSZ, CDIM, THALF, THALF, 3, 1, 1, 1, FLAG_RELU_OUT);
    cur = D1a; alt = D1b;
    for (int i = 0; i < 3; ++i) {
        int dil = 1; for (int j = 0; j < 2 - i; ++j) dil *= 3;   // 9,3,1
        run_conv(stream, pd_rw1[i], cur, d_rb1[i], nullptr, DT1, nullptr,
                 DWID, BSZ, DWID, THALF, THALF, 3, 1, dil, dil, FLAG_RELU_IN);
        run_conv(stream, pd_rw2[i], DT1, d_rb2[i], cur, alt, nullptr,
                 DWID, BSZ, DWID, THALF, THALF, 1, 1, 0, 1, FLAG_RELU_IN);
        _Float16* tmp = cur; cur = alt; alt = tmp;
    }
    // nearest-neighbor x2 upsample folded into the im2col of w_up
    run_conv(stream, pd_w_up, cur, de_b_up, nullptr, D2, nullptr,
             DWID, BSZ, DWID, THALF, TFULL, 3, 1, 1, 1, FLAG_UPSAMPLE);
    run_conv(stream, pd_w_mid, D2, de_b_mid, nullptr, D3, nullptr,
             DWID, BSZ, DWID, TFULL, TFULL, 3, 1, 1, 1, FLAG_RELU_OUT);
    run_conv(stream, pd_w_out, D3, de_b_out, nullptr, nullptr, out_f,
             BODY, BSZ, DWID, TFULL, TFULL, 3, 1, 1, 1, 0);

    // ---- scalar outputs: commit loss, perplexity ----
    vq_finalize<<<1, 256, 0, stream>>>(CNT, CACC, out_f + OUTN, out_f + OUTN + 1,
                                       NQUANT, NBCODE,
                                       1.0f / (float)(BSZ * THALF),
                                       1.0f / (float)((size_t)BSZ * THALF * CDIM));
}
